// QuantizedLinear_38122129719936
// MI455X (gfx1250) — compile-verified, matrix operand order checked by reading, not run on hardware
//
#include <hip/hip_runtime.h>

typedef __attribute__((ext_vector_type(16))) _Float16 v16h;
typedef __attribute__((ext_vector_type(8)))  float    v8f;
typedef __attribute__((ext_vector_type(8)))  _Float16 h8;
typedef __attribute__((ext_vector_type(4)))  float    f32x4;
typedef __attribute__((ext_vector_type(4)))  int      i32x4;
typedef __attribute__((ext_vector_type(4)))  _Float16 h4;

#define IN_F   4096
#define OUT_F  14336
#define TOKENS 256
#define KSTEPS (IN_F / 32)   // 128 WMMA k-steps
#define NWG    64            // N columns per workgroup (4 B-tiles)
#define MWG    128           // M rows per workgroup (4 waves' strips of 32)
#define KC     64            // K chunk staged in LDS (2 WMMA k-steps)
#define NCHUNK (IN_F / KC)   // 64

static __device__ inline v16h mk16(h4 a, h4 b, h4 c, h4 d) {
    v16h v;
#pragma unroll
    for (int i = 0; i < 4; ++i) { v[i] = a[i]; v[4+i] = b[i]; v[8+i] = c[i]; v[12+i] = d[i]; }
    return v;
}
static __device__ inline v16h mk16h8(h8 lo, h8 hi) {
    v16h v;
#pragma unroll
    for (int i = 0; i < 8; ++i) { v[i] = lo[i]; v[8+i] = hi[i]; }
    return v;
}

// ---------------------------------------------------------------------------
// Pass 1: x f32 -> f16, pre-swizzled into the WMMA 16-bit A-operand layout.
// Slot (mt, ks, lane): rows mt*16 + (lane&15), K = ks*32 + (lane>>4)*8 +
// {0..7, 16..23}. One thread per 32-byte slot -> fully coalesced stores.
// ---------------------------------------------------------------------------
__global__ __launch_bounds__(256) void x_swizzle_f16(const float* __restrict__ x,
                                                     _Float16* __restrict__ xs) {
    const int gid  = blockIdx.x * 256 + threadIdx.x;   // 65536 threads total
    const int lane = gid & 31;
    const int slot = gid >> 5;                          // mt*KSTEPS + ks
    const int ks   = slot & (KSTEPS - 1);
    const int mt   = slot >> 7;
    const int mn   = lane & 15;
    const int kb   = (lane >> 4) * 8;

    const float* src = x + (size_t)(mt * 16 + mn) * IN_F + ks * 32 + kb;
    f32x4 a0 = *(const f32x4*)(src);
    f32x4 a1 = *(const f32x4*)(src + 4);
    f32x4 a2 = *(const f32x4*)(src + 16);
    f32x4 a3 = *(const f32x4*)(src + 20);
    h4 c0 = __builtin_convertvector(a0, h4);
    h4 c1 = __builtin_convertvector(a1, h4);
    h4 c2 = __builtin_convertvector(a2, h4);
    h4 c3 = __builtin_convertvector(a3, h4);
    h8 lo, hi;
#pragma unroll
    for (int i = 0; i < 4; ++i) { lo[i] = c0[i]; lo[4+i] = c1[i]; hi[i] = c2[i]; hi[4+i] = c3[i]; }
    _Float16* dst = xs + (size_t)gid * 16;
    *(h8*)dst       = lo;
    *(h8*)(dst + 8) = hi;
}

// ---------------------------------------------------------------------------
// Pass 2: GEMM. Workgroup = 8 waves covering N=64 x M=128.
// Wave wv: mstrip = wv&3 (M base, 2 A-tiles), nhalf = wv>>2 (2 B-tiles).
// Weights cooperatively loaded/converted ONCE per chunk into LDS in the
// B-operand per-lane layout (direct GEP -> ds_store/ds_load, no flat).
// ---------------------------------------------------------------------------
template <bool AWS>
__global__ __launch_bounds__(256) void qlinear_wmma_gemm(
    const float*    __restrict__ x,     // used when !AWS
    const _Float16* __restrict__ xs,    // swizzled f16 x, used when AWS
    const int*      __restrict__ w,
    const float*    __restrict__ scale,
    const float*    __restrict__ bias,
    float*          __restrict__ y)
{
    __shared__ _Float16 lds[2][NWG * KC];   // 2 x 8 KB ping-pong

    const int tid    = threadIdx.x;
    const int lane   = tid & 31;
    const int wv     = tid >> 5;         // 0..7
    const int mstrip = wv & 3;           // 0..3  -> M strip of 32 (2 A-tiles)
    const int nhalf  = wv >> 2;          // 0..1  -> 2 B-tiles
    const int mn     = lane & 15;
    const int kb     = (lane >> 4) * 8;

    const int nBase = blockIdx.x * NWG;
    const int mBase = blockIdx.y * MWG + mstrip * 32;
    const int mt0   = mBase >> 4;        // first of 2 A tiles for this wave
    const int nt0   = nhalf * 2;         // first of 2 B tiles for this wave

    // Cooperative fill mapping: thread -> one 32B B-operand slot (tile, ks, lane)
    const int f_nt   = tid >> 6;         // 0..3
    const int f_ks   = (tid >> 5) & 1;   // 0..1
    const int f_lane = tid & 31;
    const int f_mn   = f_lane & 15;
    const int f_kb   = (f_lane >> 4) * 8;
    const int* wfill = w + (size_t)(nBase + f_nt * 16 + f_mn) * IN_F + f_ks * 32 + f_kb;
    const int f_off  = ((f_nt * 2 + f_ks) * 32 + f_lane) * 16;   // halves

    v8f acc[4] = {};   // [i*2+j]: A-tile i x B-tile j

    auto fill = [&](int buf, int k0) {
        i32x4 b0 = *(const i32x4*)(wfill + k0);
        i32x4 b1 = *(const i32x4*)(wfill + k0 + 4);
        i32x4 b2 = *(const i32x4*)(wfill + k0 + 16);
        i32x4 b3 = *(const i32x4*)(wfill + k0 + 20);
        h4 c0 = __builtin_convertvector(b0, h4);
        h4 c1 = __builtin_convertvector(b1, h4);
        h4 c2 = __builtin_convertvector(b2, h4);
        h4 c3 = __builtin_convertvector(b3, h4);
        h8 lo, hi;
#pragma unroll
        for (int i = 0; i < 4; ++i) { lo[i] = c0[i]; lo[4+i] = c1[i]; hi[i] = c2[i]; hi[4+i] = c3[i]; }
        // direct GEP on the shared array keeps addrspace(3) -> ds_store_b128
        *(h8*)&lds[buf][f_off]     = lo;
        *(h8*)&lds[buf][f_off + 8] = hi;
    };

    auto compute = [&](int buf, int kc0) {
#pragma unroll
        for (int ks = 0; ks < KC / 32; ++ks) {
            const int ksg = (kc0 >> 5) + ks;
            // Preload ALL operands of this k-step, then issue the 4 WMMAs:
            // one load clause + one wait instead of a wait per WMMA.
            v16h Bt[2];
#pragma unroll
            for (int j = 0; j < 2; ++j) {
                const int so = (((nt0 + j) * 2 + ks) * 32 + lane) * 16;
                h8 blo = *(const h8*)&lds[buf][so];       // ds_load_b128 x2
                h8 bhi = *(const h8*)&lds[buf][so + 8];
                Bt[j] = mk16h8(blo, bhi);
            }
            v16h At[2];
#pragma unroll
            for (int i = 0; i < 2; ++i) {
                if (AWS) {
                    const _Float16* asrc =
                        xs + ((size_t)((mt0 + i) * KSTEPS + ksg) * 32 + lane) * 16;
                    h8 alo = *(const h8*)asrc;            // coalesced global b128 x2
                    h8 ahi = *(const h8*)(asrc + 8);
                    At[i] = mk16h8(alo, ahi);
                } else {
                    const float* asrc =
                        x + (size_t)((mt0 + i) * 16 + mn) * IN_F + ksg * 32 + kb;
                    f32x4 a0 = *(const f32x4*)(asrc);
                    f32x4 a1 = *(const f32x4*)(asrc + 4);
                    f32x4 a2 = *(const f32x4*)(asrc + 16);
                    f32x4 a3 = *(const f32x4*)(asrc + 20);
                    At[i] = mk16(__builtin_convertvector(a0, h4), __builtin_convertvector(a1, h4),
                                 __builtin_convertvector(a2, h4), __builtin_convertvector(a3, h4));
                }
            }
#pragma unroll
            for (int i = 0; i < 2; ++i)
#pragma unroll
                for (int j = 0; j < 2; ++j)
                    acc[i * 2 + j] = __builtin_amdgcn_wmma_f32_16x16x32_f16(
                        false, At[i], false, Bt[j], (short)0, acc[i * 2 + j], false, false);
        }
    };

    // Software pipeline: one barrier per chunk (fill c+1 overlaps compute c).
    fill(0, 0);
    __syncthreads();
    for (int c = 0; c < NCHUNK; ++c) {
        if (c + 1 < NCHUNK) fill((c + 1) & 1, (c + 1) * KC);
        compute(c & 1, c * KC);
        __syncthreads();
    }

    // Epilogue: D layout VGPR r -> M = r (+8 upper lanes), N = lane&15.
    const int mOff = mBase + (lane >> 4) * 8;
#pragma unroll
    for (int j = 0; j < 2; ++j) {
        const int   o = nBase + (nt0 + j) * 16 + mn;
        const float s = scale[o];
        const float b = bias[o];
#pragma unroll
        for (int i = 0; i < 2; ++i)
#pragma unroll
            for (int r = 0; r < 8; ++r)
                y[(size_t)(mOff + i * 16 + r) * OUT_F + o] = acc[i * 2 + j][r] * s + b;
    }
}

extern "C" void kernel_launch(void* const* d_in, const int* in_sizes, int n_in,
                              void* d_out, int out_size, void* d_ws, size_t ws_size,
                              hipStream_t stream) {
    const float* x     = (const float*)d_in[0];
    const int*   w     = (const int*)d_in[1];
    const float* scale = (const float*)d_in[2];
    const float* bias  = (const float*)d_in[3];
    float*       y     = (float*)d_out;

    const size_t xs_bytes = (size_t)TOKENS * IN_F * sizeof(_Float16);  // 2 MB
    dim3 grid(OUT_F / NWG, TOKENS / MWG);   // (224, 2)
    dim3 block(256);                        // 8 waves

    if (d_ws && ws_size >= xs_bytes) {
        _Float16* xs = (_Float16*)d_ws;
        const int pre_threads = (TOKENS * IN_F) / 16;   // one thread per 32B slot
        hipLaunchKernelGGL(x_swizzle_f16, dim3(pre_threads / 256), dim3(256), 0, stream, x, xs);
        hipLaunchKernelGGL((qlinear_wmma_gemm<true>), grid, block, 0, stream,
                           x, xs, w, scale, bias, y);
    } else {
        hipLaunchKernelGGL((qlinear_wmma_gemm<false>), grid, block, 0, stream,
                           x, (const _Float16*)nullptr, w, scale, bias, y);
    }
}
